// QuantizedLinear_43602507989630
// MI455X (gfx1250) — 8-bit matrix operand order repaired and checked by reading, compiled, not run on hardware
//
#include <hip/hip_runtime.h>
#include <hip/hip_bf16.h>
#include <stdint.h>


typedef int v8i __attribute__((ext_vector_type(8)));

#define BM 128
#define BN 256
#define KS 64
#define THREADS 256

__global__ void pack_to_bytes_kernel(const int* __restrict__ src,
                                     unsigned int* __restrict__ dst,
                                     long long n4) {
    long long i = (long long)blockIdx.x * blockDim.x + threadIdx.x;
    if (i < n4) {
        const int4 v = ((const int4*)src)[i];
        unsigned int p = (unsigned int)(v.x & 255)
                       | ((unsigned int)(v.y & 255) << 8)
                       | ((unsigned int)(v.z & 255) << 16)
                       | ((unsigned int)(v.w & 255) << 24);
        dst[i] = p;
    }
}

__global__ void rowsum_kernel(const int* __restrict__ w,
                              int* __restrict__ colsum, int K) {
    __shared__ int red[THREADS];
    const int n = blockIdx.x;
    const int t = threadIdx.x;
    int s = 0;
    for (int k = t; k < K; k += THREADS) s += w[(size_t)n * K + k];
    red[t] = s;
    __syncthreads();
    for (int off = THREADS / 2; off > 0; off >>= 1) {
        if (t < off) red[t] += red[t + off];
        __syncthreads();
    }
    if (t == 0) colsum[n] = red[0];
}

__global__ __launch_bounds__(THREADS)
void qlinear_wmma_kernel(const unsigned char* __restrict__ pX,
                         const unsigned char* __restrict__ pW,
                         const int* __restrict__ colsum,
                         const float* __restrict__ bias,
                         const float* __restrict__ xs,
                         const float* __restrict__ wsc,
                         const float* __restrict__ osc,
                         const int* __restrict__ zxp,
                         const int* __restrict__ zop,
                         int* __restrict__ out,
                         int M, int N, int K) {
    __shared__ __align__(16) unsigned char Ash[2][BM][KS];
    __shared__ __align__(16) unsigned char Bsh[2][BN][KS];

    const int t      = threadIdx.x;
    const int lane   = t & 31;
    const int lhi    = lane >> 4;
    const int llo    = lane & 15;
    const int waveId = t >> 5;
    const int wave_m = waveId >> 2;
    const int wave_n = waveId & 3;

    const int blockM = blockIdx.y * BM;
    const int blockN = blockIdx.x * BN;

    const int ldRowA = t >> 1;
    const int ldColA = (t & 1) * 32;
    const unsigned char* gA = pX + (size_t)(blockM + ldRowA) * K + ldColA;
    const unsigned char* gW = pW + (size_t)(blockN + t) * K;

    v8i acc[4][4];
#pragma unroll
    for (int i = 0; i < 4; ++i)
#pragma unroll
        for (int j = 0; j < 4; ++j)
            acc[i][j] = (v8i){0, 0, 0, 0, 0, 0, 0, 0};

    const int KT = K / KS;

    int4 pa0 = *(const int4*)(gA);
    int4 pa1 = *(const int4*)(gA + 16);
    int4 pw0 = *(const int4*)(gW);
    int4 pw1 = *(const int4*)(gW + 16);
    int4 pw2 = *(const int4*)(gW + 32);
    int4 pw3 = *(const int4*)(gW + 48);
    *(int4*)&Ash[0][ldRowA][ldColA]      = pa0;
    *(int4*)&Ash[0][ldRowA][ldColA + 16] = pa1;
    *(int4*)&Bsh[0][t][0]  = pw0;
    *(int4*)&Bsh[0][t][16] = pw1;
    *(int4*)&Bsh[0][t][32] = pw2;
    *(int4*)&Bsh[0][t][48] = pw3;
    __syncthreads();

    for (int kt = 0; kt < KT; ++kt) {
        const int buf = kt & 1;

        if (kt + 1 < KT) {
            const size_t off = (size_t)(kt + 1) * KS;
            pa0 = *(const int4*)(gA + off);
            pa1 = *(const int4*)(gA + off + 16);
            pw0 = *(const int4*)(gW + off);
            pw1 = *(const int4*)(gW + off + 16);
            pw2 = *(const int4*)(gW + off + 32);
            pw3 = *(const int4*)(gW + off + 48);
        }

        v8i bfrag[4];
#pragma unroll
        for (int j = 0; j < 4; ++j) {
            const unsigned char* bp =
                &Bsh[buf][wave_n * 64 + j * 16 + llo][lhi * 16];
            int4 b0 = *(const int4*)(bp);
            int4 b1 = *(const int4*)(bp + 32);
            bfrag[j] = (v8i){b0.x, b0.y, b0.z, b0.w, b1.x, b1.y, b1.z, b1.w};
        }

#pragma unroll
        for (int i = 0; i < 4; ++i) {
            const unsigned char* ap =
                &Ash[buf][wave_m * 64 + i * 16 + llo][lhi * 16];
            int2 a0 = *(const int2*)(ap);
            int2 a1 = *(const int2*)(ap + 8);
            int2 a2 = *(const int2*)(ap + 32);
            int2 a3 = *(const int2*)(ap + 40);
            v8i afrag = (v8i){a0.x, a0.y, a1.x, a1.y, a2.x, a2.y, a3.x, a3.y};
#pragma unroll
            for (int j = 0; j < 4; ++j)
                acc[i][j] = __builtin_amdgcn_wmma_i32_16x16x64_iu8(
                     false, afrag,
                     true,  bfrag[j],
                    acc[i][j],
                     false,  false);
        }

        if (kt + 1 < KT) {
            const int nbuf = buf ^ 1;
            *(int4*)&Ash[nbuf][ldRowA][ldColA]      = pa0;
            *(int4*)&Ash[nbuf][ldRowA][ldColA + 16] = pa1;
            *(int4*)&Bsh[nbuf][t][0]  = pw0;
            *(int4*)&Bsh[nbuf][t][16] = pw1;
            *(int4*)&Bsh[nbuf][t][32] = pw2;
            *(int4*)&Bsh[nbuf][t][48] = pw3;
        }
        __syncthreads();
    }

    const float s      = xs[0] * wsc[0];
    const float invOut = 1.0f / osc[0];
    const int   zpx    = zxp[0];
    const int   zpo    = zop[0];

#pragma unroll
    for (int j = 0; j < 4; ++j) {
        const int n  = blockN + wave_n * 64 + j * 16 + llo;
        const int cs = colsum[n];
        const float bv = bias[n];
#pragma unroll
        for (int i = 0; i < 4; ++i) {
            const int mbase = blockM + wave_m * 64 + i * 16 + lhi * 8;
#pragma unroll
            for (int e = 0; e < 8; ++e) {
                const int av = acc[i][j][e] - zpx * cs;
                const float y = (float)av * s + bv;
                int q = (int)rintf(y * invOut) + zpo;
                q = q < 0 ? 0 : (q > 255 ? 255 : q);
                out[(size_t)(mbase + e) * N + n] = q;
            }
        }
    }
}

extern "C" void kernel_launch(void* const* d_in, const int* in_sizes, int n_in,
                              void* d_out, int out_size, void* d_ws, size_t ws_size,
                              hipStream_t stream) {
    const int*   x_q  = (const int*)d_in[0];
    const int*   w_q  = (const int*)d_in[1];
    const float* bias = (const float*)d_in[2];
    const float* xs   = (const float*)d_in[3];
    const float* wsc  = (const float*)d_in[4];
    const float* osc  = (const float*)d_in[5];
    const int*   zx   = (const int*)d_in[6];
    const int*   zo   = (const int*)d_in[7];
    int* out = (int*)d_out;

    const int N = in_sizes[2];
    const int K = in_sizes[1] / N;
    const int M = in_sizes[0] / K;

    unsigned char* ws = (unsigned char*)d_ws;
    unsigned char* packedX = ws;
    unsigned char* packedW = ws + (size_t)M * K;
    int* colsum = (int*)(packedW + (size_t)N * K);

    {
        long long n4 = ((long long)M * K) / 4;
        long long blocks = (n4 + THREADS - 1) / THREADS;
        pack_to_bytes_kernel<<<(unsigned)blocks, THREADS, 0, stream>>>(
            x_q, (unsigned int*)packedX, n4);
    }
    {
        long long n4 = ((long long)N * K) / 4;
        long long blocks = (n4 + THREADS - 1) / THREADS;
        pack_to_bytes_kernel<<<(unsigned)blocks, THREADS, 0, stream>>>(
            w_q, (unsigned int*)packedW, n4);
    }
    rowsum_kernel<<<N, THREADS, 0, stream>>>(w_q, colsum, K);

    dim3 grid(N / BN, M / BM);
    qlinear_wmma_kernel<<<grid, THREADS, 0, stream>>>(
        packedX, packedW, colsum, bias, xs, wsc, osc, zx, zo, out, M, N, K);
}
